// GINEEdgeModel_23519240913054
// MI455X (gfx1250) — compile-verified, matrix operand order checked by reading, not run on hardware
//
#include <hip/hip_runtime.h>
#include <hip/hip_bf16.h>

typedef _Float16 f16;
typedef _Float16 v16h __attribute__((ext_vector_type(16)));
typedef float    v8f  __attribute__((ext_vector_type(8)));

#define N_NODES_C 100000
#define N_EDGES_C 600000
#define NODE_IN_C 387
#define HID_C     128
#define EDGE_DIM_C 7
#define KPAD1 416   // 387 padded to 13*32
#define KPADE 288   // 263 padded to 9*32

// ---------------- WMMA fragment helpers (wave32, 16x16x32 f16) ----------------
// A (16x32 f16, row-major LDS tile): lanes 0-15 hold row M=lane with K {0..7,16..23},
// lanes 16-31 hold row M=lane-16 with K {8..15,24..31}.
__device__ __forceinline__ v16h load_frag_row(const f16* p, int ld, int r, int k0, int h) {
  v16h v;
  const f16* base = p + r * ld + k0 + h * 8;
#pragma unroll
  for (int i = 0; i < 8; ++i) v[i] = base[i];
#pragma unroll
  for (int i = 0; i < 8; ++i) v[8 + i] = base[16 + i];
  return v;
}

// B fragment from pre-packed weights: the 16 halves lane (col, h) needs for K-step
// `kstep` are contiguous (32B) -> lowers to wide coalesced global loads.
__device__ __forceinline__ v16h load_frag_packed(const f16* p, int kstep, int col, int h) {
  return *(const v16h*)(p + ((((kstep * HID_C) + col) * 2 + h) << 4));
}

// ---------------- weight convert + zero-pad + fragment-pack  ----------------
// src: f32 [K][128] row-major.  dst: f16 packed [KP/32][128 cols][2 halves][16].
__global__ void convert_pack(const float* __restrict__ src, f16* __restrict__ dst,
                             int K, int KP) {
  int idx = blockIdx.x * 256 + threadIdx.x;
  if (idx >= KP * HID_C) return;
  int k = idx >> 7;          // row
  int c = idx & 127;         // col
  int kstep = k >> 5;
  int kk = k & 31;
  int h = (kk >> 3) & 1;                       // which lane-half wants this k
  int i = (kk & 7) + ((kk & 16) ? 8 : 0);      // position within the 16-half fragment
  f16 v = (k < K) ? (f16)src[k * HID_C + c] : (f16)0.f;
  dst[((((kstep * HID_C) + c) * 2 + h) << 4) + i] = v;
}

// ---------------- fused edge-linear + relu + atomic scatter ----------------
// msg = relu(x[src] + ea @ ew + eb); aggr[dst] += msg.  2 edges per 256-thread block.
__global__ void edge_scatter(const float* __restrict__ x, const int* __restrict__ src,
                             const int* __restrict__ dst, const float* __restrict__ ea,
                             const float* __restrict__ ew, const float* __restrict__ eb,
                             float* __restrict__ aggr, int n_edges, int dim) {
  int eloc = threadIdx.x >> 7;
  int c0 = threadIdx.x & 127;
  int e = blockIdx.x * 2 + eloc;
  if (e >= n_edges) return;
  int s = src[e], d = dst[e];
  float a[EDGE_DIM_C];
#pragma unroll
  for (int k = 0; k < EDGE_DIM_C; ++k) a[k] = ea[(long)e * EDGE_DIM_C + k];
  const float* xrow = x + (long)s * dim;
  float* arow = aggr + (long)d * dim;
  for (int c = c0; c < dim; c += 128) {
    float el = eb[c];
#pragma unroll
    for (int k = 0; k < EDGE_DIM_C; ++k) el += a[k] * ew[k * dim + c];
    float m = xrow[c] + el;
    if (m > 0.f)  // relu: zero messages contribute nothing -> skip the atomic
      (void)__hip_atomic_fetch_add(&arow[c], m, __ATOMIC_RELAXED, __HIP_MEMORY_SCOPE_AGENT);
  }
}

// ---------------- fused node MLP: relu(relu((x+aggr)@W1+b1)@W2+b2) ----------------
// 16 nodes per 256-thread block (8 waves); each wave owns a 16-col slice of N=128.
template <int KPAD>
__global__ void node_mlp(const float* __restrict__ xin, const float* __restrict__ aggr,
                         const f16* __restrict__ w1p, const float* __restrict__ b1,
                         const f16* __restrict__ w2p, const float* __restrict__ b2,
                         float* __restrict__ out, int n_nodes, int kin) {
  __shared__ f16 As[16 * KPAD];
  __shared__ f16 Hs[16 * HID_C];
  int tid = threadIdx.x;
  int node0 = blockIdx.x * 16;

  for (int idx = tid; idx < 16 * KPAD; idx += 256) {
    int row = idx / KPAD;
    int k = idx - row * KPAD;
    float v = 0.f;
    int n = node0 + row;
    if (k < kin && n < n_nodes) {
      long o = (long)n * kin + k;
      v = xin[o] + aggr[o];
    }
    As[idx] = (f16)v;
  }
  __syncthreads();

  int wv = tid >> 5, lane = tid & 31;
  int h = lane >> 4, r = lane & 15;
  int gcol = wv * 16 + r;

  v8f acc = {};
#pragma unroll
  for (int k0 = 0; k0 < KPAD; k0 += 32) {
    v16h a = load_frag_row(As, KPAD, r, k0, h);
    v16h b = load_frag_packed(w1p, k0 >> 5, gcol, h);
    acc = __builtin_amdgcn_wmma_f32_16x16x32_f16(false, a, false, b, (short)0, acc, false, false);
  }
  float bias1 = b1[gcol];
#pragma unroll
  for (int j = 0; j < 8; ++j) {
    int m = h * 8 + j;
    float v = acc[j] + bias1;
    Hs[m * HID_C + gcol] = (f16)(v > 0.f ? v : 0.f);
  }
  __syncthreads();

  v8f acc2 = {};
#pragma unroll
  for (int k0 = 0; k0 < HID_C; k0 += 32) {
    v16h a = load_frag_row(Hs, HID_C, r, k0, h);
    v16h b = load_frag_packed(w2p, k0 >> 5, gcol, h);
    acc2 = __builtin_amdgcn_wmma_f32_16x16x32_f16(false, a, false, b, (short)0, acc2, false, false);
  }
  float bias2 = b2[gcol];
#pragma unroll
  for (int j = 0; j < 8; ++j) {
    int m = h * 8 + j;
    int n = node0 + m;
    if (n < n_nodes) {
      float v = acc2[j] + bias2;
      out[(long)n * HID_C + gcol] = v > 0.f ? v : 0.f;  // post-conv relu fused
    }
  }
}

// ---------------- edge head: relu([h2[src]|h2[dst]|ea] @ W1 + b1) @ W2 + b2 ----------------
__global__ void edge_head(const float* __restrict__ h2, const int* __restrict__ src,
                          const int* __restrict__ dst, const float* __restrict__ ea,
                          const f16* __restrict__ w1p, const float* __restrict__ b1,
                          const float* __restrict__ w2, const float* __restrict__ b2,
                          float* __restrict__ out, int n_edges) {
  __shared__ f16 As[16 * KPADE];
  __shared__ float Hf[16 * HID_C];
  __shared__ int sidx[16], didx[16];
  int tid = threadIdx.x;
  int e0 = blockIdx.x * 16;

  if (tid < 16) {
    int e = e0 + tid;
    sidx[tid] = (e < n_edges) ? src[e] : 0;
    didx[tid] = (e < n_edges) ? dst[e] : 0;
  }
  __syncthreads();

  for (int idx = tid; idx < 16 * KPADE; idx += 256) {
    int row = idx / KPADE;
    int col = idx - row * KPADE;
    float v = 0.f;
    int e = e0 + row;
    if (e < n_edges) {
      if (col < HID_C)               v = h2[(long)sidx[row] * HID_C + col];
      else if (col < 2 * HID_C)      v = h2[(long)didx[row] * HID_C + (col - HID_C)];
      else if (col < 2 * HID_C + EDGE_DIM_C)
                                     v = ea[(long)e * EDGE_DIM_C + (col - 2 * HID_C)];
    }
    As[idx] = (f16)v;
  }
  __syncthreads();

  int wv = tid >> 5, lane = tid & 31;
  int h = lane >> 4, r = lane & 15;
  int gcol = wv * 16 + r;

  v8f acc = {};
#pragma unroll
  for (int k0 = 0; k0 < KPADE; k0 += 32) {
    v16h a = load_frag_row(As, KPADE, r, k0, h);
    v16h b = load_frag_packed(w1p, k0 >> 5, gcol, h);
    acc = __builtin_amdgcn_wmma_f32_16x16x32_f16(false, a, false, b, (short)0, acc, false, false);
  }
  float bias1 = b1[gcol];
#pragma unroll
  for (int j = 0; j < 8; ++j) {
    int m = h * 8 + j;
    float v = acc[j] + bias1;
    Hf[m * HID_C + gcol] = v > 0.f ? v : 0.f;
  }
  __syncthreads();

  // tiny 128 -> 2 GEMM: 32 dot products, keep f32 for the final reduction
  if (tid < 32) {
    int el = tid >> 1, j = tid & 1;
    int e = e0 + el;
    if (e < n_edges) {
      float s = b2[j];
      for (int c = 0; c < HID_C; ++c) s += Hf[el * HID_C + c] * w2[c * 2 + j];
      out[(long)e * 2 + j] = s;
    }
  }
}

// ---------------- launch ----------------
extern "C" void kernel_launch(void* const* d_in, const int* in_sizes, int n_in,
                              void* d_out, int out_size, void* d_ws, size_t ws_size,
                              hipStream_t stream) {
  const float* x    = (const float*)d_in[0];
  const int*   ei   = (const int*)d_in[1];
  const float* ea   = (const float*)d_in[2];
  const float* e1w  = (const float*)d_in[3];
  const float* e1b  = (const float*)d_in[4];
  const float* m1w1 = (const float*)d_in[5];
  const float* m1b1 = (const float*)d_in[6];
  const float* m1w2 = (const float*)d_in[7];
  const float* m1b2 = (const float*)d_in[8];
  const float* e2w  = (const float*)d_in[9];
  const float* e2b  = (const float*)d_in[10];
  const float* m2w1 = (const float*)d_in[11];
  const float* m2b1 = (const float*)d_in[12];
  const float* m2w2 = (const float*)d_in[13];
  const float* m2b2 = (const float*)d_in[14];
  const float* emw1 = (const float*)d_in[15];
  const float* emb1 = (const float*)d_in[16];
  const float* emw2 = (const float*)d_in[17];
  const float* emb2 = (const float*)d_in[18];

  const int* src = ei;
  const int* dst = ei + N_EDGES_C;

  char* p = (char*)d_ws;
  auto alloc = [&](size_t bytes) {
    char* r = p;
    p += (bytes + 255) & ~(size_t)255;
    return r;
  };
  float* aggr1 = (float*)alloc((size_t)N_NODES_C * NODE_IN_C * 4);
  float* aggr2 = (float*)alloc((size_t)N_NODES_C * HID_C * 4);
  float* h1    = (float*)alloc((size_t)N_NODES_C * HID_C * 4);
  float* h2    = (float*)alloc((size_t)N_NODES_C * HID_C * 4);
  f16* w1p = (f16*)alloc((size_t)KPAD1 * HID_C * 2);
  f16* w2p = (f16*)alloc((size_t)HID_C * HID_C * 2);
  f16* w3p = (f16*)alloc((size_t)HID_C * HID_C * 2);
  f16* w4p = (f16*)alloc((size_t)HID_C * HID_C * 2);
  f16* w5p = (f16*)alloc((size_t)KPADE * HID_C * 2);

  hipMemsetAsync(aggr1, 0, (size_t)N_NODES_C * NODE_IN_C * 4, stream);
  hipMemsetAsync(aggr2, 0, (size_t)N_NODES_C * HID_C * 4, stream);

  auto cvt = [&](const float* s, f16* dptr, int K, int KP) {
    int total = KP * HID_C;
    convert_pack<<<(total + 255) / 256, 256, 0, stream>>>(s, dptr, K, KP);
  };
  cvt(m1w1, w1p, NODE_IN_C, KPAD1);
  cvt(m1w2, w2p, HID_C, HID_C);
  cvt(m2w1, w3p, HID_C, HID_C);
  cvt(m2w2, w4p, HID_C, HID_C);
  cvt(emw1, w5p, 2 * HID_C + EDGE_DIM_C, KPADE);

  // conv1: scatter then fused MLP
  edge_scatter<<<(N_EDGES_C + 1) / 2, 256, 0, stream>>>(x, src, dst, ea, e1w, e1b,
                                                        aggr1, N_EDGES_C, NODE_IN_C);
  node_mlp<KPAD1><<<(N_NODES_C + 15) / 16, 256, 0, stream>>>(x, aggr1, w1p, m1b1, w2p, m1b2,
                                                             h1, N_NODES_C, NODE_IN_C);
  // conv2
  edge_scatter<<<(N_EDGES_C + 1) / 2, 256, 0, stream>>>(h1, src, dst, ea, e2w, e2b,
                                                        aggr2, N_EDGES_C, HID_C);
  node_mlp<HID_C><<<(N_NODES_C + 15) / 16, 256, 0, stream>>>(h1, aggr2, w3p, m2b1, w4p, m2b2,
                                                             h2, N_NODES_C, HID_C);
  // edge head
  edge_head<<<(N_EDGES_C + 15) / 16, 256, 0, stream>>>(h2, src, dst, ea, w5p, emb1,
                                                       emw2, emb2, (float*)d_out, N_EDGES_C);
}
